// Layer_71554155151949
// MI455X (gfx1250) — compile-verified, matrix operand order checked by reading, not run on hardware
//
#include <hip/hip_runtime.h>

typedef __attribute__((ext_vector_type(16))) __bf16 v16bf;
typedef __attribute__((ext_vector_type(8)))  __bf16 v8bf;
typedef __attribute__((ext_vector_type(8)))  float  v8f;
typedef __attribute__((ext_vector_type(4)))  unsigned v4u;
typedef __attribute__((ext_vector_type(4)))  int  v4i;
typedef __attribute__((ext_vector_type(8)))  int  v8i;

#define DEV_ static __device__ __forceinline__

constexpr int kD    = 1024;
constexpr int kT    = 2048;
constexpr int kBATCH= 4;
constexpr int kH    = 16;
constexpr int kE    = 8;
constexpr int kHD   = 2730;
constexpr int kHDP  = 2752;            // HD padded to multiple of 64
constexpr int kNTOK = kBATCH * kT;     // 8192
constexpr int kSLOTS= 17408;           // >= 2*NTOK + 8*127, multiple of 128

// ---------------- WMMA helpers (CDNA5 16x16x32 bf16) ----------------

DEV_ v8f wmma_bf16(v16bf a, v16bf b, v8f c) {
  return __builtin_amdgcn_wmma_f32_16x16x32_bf16(false, a, false, b, (short)0, c, false, false);
}

DEV_ v8f zero8() { v8f z = {0.f,0.f,0.f,0.f,0.f,0.f,0.f,0.f}; return z; }

DEV_ v8bf zero8bf() {
  v8bf z;
#pragma unroll
  for (int i = 0; i < 8; i++) z[i] = (__bf16)0.f;
  return z;
}

DEV_ v8bf ldg8(const __bf16* p) { return *(const v8bf*)p; }

// Per-lane fragment load for 16-bit A (16x32) / B-transposed (n-major) tiles.
// `s` points at this lane's 32 contiguous bf16 K-values (16B aligned).
// Fragment element i holds K = (i>>3)*16 + half*8 + (i&7)   (ISA 7.12.2).
DEV_ v16bf ld_slice(const __bf16* s, int hf) {
  const v8bf* p = (const v8bf*)s;
  v8bf lo = p[hf];
  v8bf hi = p[2 + hf];
  v16bf r;
#pragma unroll
  for (int i = 0; i < 8; i++) { r[i] = lo[i]; r[8 + i] = hi[i]; }
  return r;
}
DEV_ v16bf ld_frag(const __bf16* tile, int row, int hf) { return ld_slice(tile + row * 32, hf); }

DEV_ float rmax16(float v) {
#pragma unroll
  for (int m = 1; m < 16; m <<= 1) v = fmaxf(v, __shfl_xor(v, m, 32));
  return v;
}
DEV_ float rsum16(float v) {
#pragma unroll
  for (int m = 1; m < 16; m <<= 1) v += __shfl_xor(v, m, 32);
  return v;
}

// ---------------- Tensor Data Mover: 2D tile (rows x 32 bf16) -> LDS ----------------
// D# layout per CDNA5 ISA 8.3/8.4.  Tensor dims == tile dims (no OOB clipping needed).

DEV_ void tdm_load_tile(unsigned lds_addr, const __bf16* gptr, int rows, int stride_elems) {
  unsigned long long ga = (unsigned long long)(uintptr_t)gptr;
  v4u g0;
  g0[0] = 1u;                                             // count=1, user mode
  g0[1] = lds_addr;                                       // lds_addr [63:32]
  g0[2] = (unsigned)ga;                                   // global_addr lo
  g0[3] = (unsigned)((ga >> 32) & 0x01ffffffu) | (2u << 30);  // global_addr hi | type=2
  v8i g1;
  g1[0] = 0x10000;                                        // data_size=1 (2 bytes)
  g1[1] = 32 << 16;                                       // tensor_dim0 = 32
  g1[2] = rows << 16;                                     // tensor_dim1 = rows
  g1[3] = 32 << 16;                                       // tile_dim0 = 32
  g1[4] = rows & 0xffff;                                  // tile_dim1 = rows
  g1[5] = stride_elems;                                   // tensor_dim0_stride lo
  g1[6] = 0;
  g1[7] = 0;
  v4i z4 = {0, 0, 0, 0};
#if defined(__clang_major__) && (__clang_major__ >= 23)
  v8i z8 = {0, 0, 0, 0, 0, 0, 0, 0};
  __builtin_amdgcn_tensor_load_to_lds(g0, g1, z4, z4, z8, 0);
#else
  __builtin_amdgcn_tensor_load_to_lds(g0, g1, z4, z4, 0);
#endif
}

// ---------------- LayerNorm -> bf16 ----------------

__global__ void ln_kernel(const float* __restrict__ x, const float* __restrict__ g,
                          const float* __restrict__ b, __bf16* __restrict__ out) {
  int t = blockIdx.x, tid = threadIdx.x;
  const float* xr = x + (size_t)t * kD;
  float v[4], s = 0.f, s2 = 0.f;
#pragma unroll
  for (int k = 0; k < 4; k++) {
    v[k] = xr[tid + k * 256];
    s += v[k];
    s2 += v[k] * v[k];
  }
  __shared__ float rs[256], rq[256];
  rs[tid] = s; rq[tid] = s2;
  __syncthreads();
  for (int st = 128; st > 0; st >>= 1) {
    if (tid < st) { rs[tid] += rs[tid + st]; rq[tid] += rq[tid + st]; }
    __syncthreads();
  }
  float mean = rs[0] / kD;
  float var  = rq[0] / kD - mean * mean;
  float inv  = rsqrtf(var + 1e-5f);
#pragma unroll
  for (int k = 0; k < 4; k++) {
    int i = tid + k * 256;
    out[(size_t)t * kD + i] = (__bf16)((v[k] - mean) * inv * g[i] + b[i]);
  }
}

// ---------------- weight pack kernels (all produce B^T [N][K] bf16) ----------------

// wq/wk/wv [H,D,HS] f32 -> BT [H*HS][D] bf16
__global__ void pack_headw_t(const float* __restrict__ in, __bf16* __restrict__ out) {
  int i = blockIdx.x * 256 + threadIdx.x;
  if (i >= kD * kD) return;
  int d = i & (kD - 1), n = i >> 10;    // out[n][d]
  int h = n >> 6, s = n & 63;
  out[i] = (__bf16)in[((size_t)h * kD + d) * 64 + s];
}

// wo [D,D] f32 -> BT [n][d] bf16
__global__ void pack_t(const float* __restrict__ in, __bf16* __restrict__ out) {
  int i = blockIdx.x * 256 + threadIdx.x;
  if (i >= kD * kD) return;
  int d = i & (kD - 1), n = i >> 10;
  out[i] = (__bf16)in[(size_t)d * kD + n];
}

// w1/w2 [E,D,HD] -> BT [E][HDP][D] bf16, zero-padded n rows
__global__ void pack_w12_t(const float* __restrict__ in, __bf16* __restrict__ out) {
  size_t i = (size_t)blockIdx.x * 256 + threadIdx.x;
  if (i >= (size_t)kE * kHDP * kD) return;
  int d = (int)(i & (kD - 1));
  size_t rr = i >> 10;
  int n = (int)(rr % kHDP);
  size_t e = rr / kHDP;
  out[i] = (n < kHD) ? (__bf16)in[((size_t)e * kD + d) * kHD + n] : (__bf16)0.f;
}

// w3 [E,HD,D] -> BT [E][D][HDP] bf16, zero-padded k cols
__global__ void pack_w3_t(const float* __restrict__ in, __bf16* __restrict__ out) {
  size_t i = (size_t)blockIdx.x * 256 + threadIdx.x;
  if (i >= (size_t)kE * kD * kHDP) return;
  int kk = (int)(i % kHDP);
  size_t rr = i / kHDP;
  int n = (int)(rr & (kD - 1));
  size_t e = rr >> 10;
  out[i] = (kk < kHD) ? (__bf16)in[((size_t)e * kHD + kk) * kD + n] : (__bf16)0.f;
}

// ---------------- 128x128 WMMA GEMM, double-buffered LDS ----------------
// C = A[M,K] x B  (B given as BT[N][K]).  Wave tile 32x64 -> 8 WMMA / k-step.
// EPI 0: bf16 [row,col].  EPI 1: f32 = acc+bias[col]+resid[row,col].
// EPI 2: bf16 transposed V store: out[(b*kD+col)*kT + t], row = b*kT+t.

template <int EPI>
__global__ void gemm128(const __bf16* __restrict__ A, const __bf16* __restrict__ BT,
                        void* __restrict__ outp, const float* __restrict__ bias,
                        const float* __restrict__ resid, int M, int N, int K) {
  __shared__ alignas(16) __bf16 At[2][128 * 32];
  __shared__ alignas(16) __bf16 Bt[2][128 * 32];
  int tid = threadIdx.x;
  int bm0 = blockIdx.y * 128, bn0 = blockIdx.x * 128;
  int lane = tid & 31, w = tid >> 5;
  int wm = (w & 3) * 32, wn = (w >> 2) * 64;
  int ln = lane & 15, hf = lane >> 4;
  int arow = tid >> 1, ak = (tid & 1) * 16;

  const __bf16* Ap = A + (size_t)(bm0 + arow) * K + ak;
  const __bf16* Bp = BT + (size_t)(bn0 + arow) * K + ak;
  bool aval = (bm0 + arow) < M;

  v8f acc[2][4];
#pragma unroll
  for (int i = 0; i < 2; i++)
#pragma unroll
    for (int j = 0; j < 4; j++) acc[i][j] = zero8();

  int nt = K / 32;
  v8bf ag0 = aval ? ldg8(Ap) : zero8bf();
  v8bf ag1 = aval ? ldg8(Ap + 8) : zero8bf();
  v8bf bg0 = ldg8(Bp), bg1 = ldg8(Bp + 8);
  { __bf16* d0 = At[0] + arow * 32 + ak; *(v8bf*)d0 = ag0; *(v8bf*)(d0 + 8) = ag1;
    __bf16* d1 = Bt[0] + arow * 32 + ak; *(v8bf*)d1 = bg0; *(v8bf*)(d1 + 8) = bg1; }
  __syncthreads();

  int buf = 0;
  for (int kt = 0; kt < nt; kt++) {
    if (kt + 1 < nt) {
      const __bf16* An = Ap + (kt + 1) * 32;
      const __bf16* Bn = Bp + (kt + 1) * 32;
      ag0 = aval ? ldg8(An) : zero8bf();
      ag1 = aval ? ldg8(An + 8) : zero8bf();
      bg0 = ldg8(Bn); bg1 = ldg8(Bn + 8);
      if (kt + 2 < nt) {
        __builtin_prefetch(Ap + (kt + 2) * 32, 0, 0);
        __builtin_prefetch(Bp + (kt + 2) * 32, 0, 0);
      }
    }
    const __bf16* Ab = At[buf];
    const __bf16* Bb = Bt[buf];
    v16bf af0 = ld_frag(Ab, wm + ln, hf);
    v16bf af1 = ld_frag(Ab, wm + 16 + ln, hf);
#pragma unroll
    for (int j = 0; j < 4; j++) {
      v16bf bfj = ld_frag(Bb, wn + j * 16 + ln, hf);
      acc[0][j] = wmma_bf16(af0, bfj, acc[0][j]);
      acc[1][j] = wmma_bf16(af1, bfj, acc[1][j]);
    }
    if (kt + 1 < nt) {
      __bf16* d0 = At[buf ^ 1] + arow * 32 + ak; *(v8bf*)d0 = ag0; *(v8bf*)(d0 + 8) = ag1;
      __bf16* d1 = Bt[buf ^ 1] + arow * 32 + ak; *(v8bf*)d1 = bg0; *(v8bf*)(d1 + 8) = bg1;
    }
    __syncthreads();
    buf ^= 1;
  }

#pragma unroll
  for (int i = 0; i < 2; i++)
#pragma unroll
    for (int j = 0; j < 4; j++)
#pragma unroll
      for (int r = 0; r < 8; r++) {
        int row = bm0 + wm + i * 16 + r + 8 * hf;
        if (row >= M) continue;
        int col = bn0 + wn + j * 16 + ln;
        float v = acc[i][j][r];
        if (EPI == 0) {
          ((__bf16*)outp)[(size_t)row * N + col] = (__bf16)v;
        } else if (EPI == 1) {
          ((float*)outp)[(size_t)row * N + col] =
              v + bias[col] + resid[(size_t)row * N + col];
        } else {  // EPI 2: transposed V store
          int bb = row >> 11, t = row & (kT - 1);
          ((__bf16*)outp)[((size_t)bb * kD + col) * kT + t] = (__bf16)v;
        }
      }
}

// ---------------- flash attention (causal, per (b,h)) ----------------
// Block: 256 threads = 8 waves; wave owns 16 query rows; block covers 128.
// V is pre-transposed: Vt[(b*kD + h*64 + hs)*kT + t].

__global__ void flash_attn(const __bf16* __restrict__ Q, const __bf16* __restrict__ Kb,
                           const __bf16* __restrict__ Vt, __bf16* __restrict__ O) {
  int bh = blockIdx.y;
  int b = bh >> 4, h = bh & 15;
  int qb0 = blockIdx.x * 128;
  int tid = threadIdx.x, w = tid >> 5, lane = tid & 31;
  int ln = lane & 15, hf = lane >> 4;
  int q0w = qb0 + w * 16;
  const float NEG_INF = -__builtin_inff();

  __shared__ alignas(16) __bf16 Pb[2][8 * 512];  // double-buffered per-wave P tiles

  size_t baseRow = ((size_t)b * kT) * kD + (size_t)h * 64;
  size_t vbase = ((size_t)b * kD + h * 64) * kT;

  v16bf qf0 = ld_slice(Q + baseRow + (size_t)(q0w + ln) * kD + 0,  hf);
  v16bf qf1 = ld_slice(Q + baseRow + (size_t)(q0w + ln) * kD + 32, hf);

  v8f o0 = zero8(), o1 = zero8(), o2 = zero8(), o3 = zero8();
  float mrow[8], lrow[8];
#pragma unroll
  for (int r = 0; r < 8; r++) { mrow[r] = NEG_INF; lrow[r] = 0.f; }

  int KB = (qb0 + 128) >> 5;
  for (int kb = 0; kb < KB; kb++) {
    int k0 = kb * 32;
    bool active = (k0 <= q0w + 15);
    __bf16* Pw = Pb[kb & 1] + w * 512;
    if (active) {
      const __bf16* kp0 = Kb + baseRow + (size_t)(k0 + ln) * kD;
      const __bf16* kp1 = Kb + baseRow + (size_t)(k0 + 16 + ln) * kD;
      v16bf ka0 = ld_slice(kp0 + 0, hf),  ka1 = ld_slice(kp0 + 32, hf);
      v16bf kc0 = ld_slice(kp1 + 0, hf),  kc1 = ld_slice(kp1 + 32, hf);
      v8f s0 = zero8(), s1 = zero8();
      s0 = wmma_bf16(qf0, ka0, s0); s0 = wmma_bf16(qf1, ka1, s0);
      s1 = wmma_bf16(qf0, kc0, s1); s1 = wmma_bf16(qf1, kc1, s1);
#pragma unroll
      for (int r = 0; r < 8; r++) {
        int q = q0w + r + 8 * hf;
        float a0 = s0[r] * 0.125f; if (k0 + ln > q)      a0 = NEG_INF;
        float a1 = s1[r] * 0.125f; if (k0 + 16 + ln > q) a1 = NEG_INF;
        float tmax = rmax16(fmaxf(a0, a1));
        float nm = fmaxf(mrow[r], tmax);
        float corr = __expf(mrow[r] - nm);
        float p0 = __expf(a0 - nm);
        float p1 = __expf(a1 - nm);
        int m = r + 8 * hf;
        Pw[m * 32 + ln]      = (__bf16)p0;
        Pw[m * 32 + 16 + ln] = (__bf16)p1;
        lrow[r] = lrow[r] * corr + rsum16(p0 + p1);
        mrow[r] = nm;
        o0[r] *= corr; o1[r] *= corr; o2[r] *= corr; o3[r] *= corr;
      }
    }
    __syncthreads();
    if (active) {
      v16bf pf = ld_frag(Pw, ln, hf);
      v16bf v0 = ld_slice(Vt + vbase + (size_t)(0  + ln) * kT + k0, hf);
      v16bf v1 = ld_slice(Vt + vbase + (size_t)(16 + ln) * kT + k0, hf);
      v16bf v2 = ld_slice(Vt + vbase + (size_t)(32 + ln) * kT + k0, hf);
      v16bf v3 = ld_slice(Vt + vbase + (size_t)(48 + ln) * kT + k0, hf);
      o0 = wmma_bf16(pf, v0, o0);
      o1 = wmma_bf16(pf, v1, o1);
      o2 = wmma_bf16(pf, v2, o2);
      o3 = wmma_bf16(pf, v3, o3);
    }
  }
#pragma unroll
  for (int r = 0; r < 8; r++) {
    int q = q0w + r + 8 * hf;
    float inv = 1.f / lrow[r];
    size_t rb = baseRow + (size_t)q * kD;
    O[rb + 0 + ln]  = (__bf16)(o0[r] * inv);
    O[rb + 16 + ln] = (__bf16)(o1[r] * inv);
    O[rb + 32 + ln] = (__bf16)(o2[r] * inv);
    O[rb + 48 + ln] = (__bf16)(o3[r] * inv);
  }
}

// ---------------- MoE routing ----------------

__global__ void moe_zero(int* counts, int* fill) {
  int i = threadIdx.x;
  if (i < kE) { counts[i] = 0; fill[i] = 0; }
}

__global__ void slot_init(int* slot_token) {
  int i = blockIdx.x * 256 + threadIdx.x;
  if (i < kSLOTS) slot_token[i] = -1;
}

__global__ void router_kernel(const __bf16* __restrict__ xn, const float* __restrict__ rw,
                              const float* __restrict__ rb, int* __restrict__ tok_e,
                              float* __restrict__ tok_g, int* __restrict__ counts) {
  int i = blockIdx.x * 256 + threadIdx.x;
  if (i >= kNTOK) return;
  float acc[kE];
#pragma unroll
  for (int e = 0; e < kE; e++) acc[e] = rb[e];
  const __bf16* xr = xn + (size_t)i * kD;
  for (int d = 0; d < kD; d++) {
    float xv = (float)xr[d];
    const float* wr = rw + d * kE;
#pragma unroll
    for (int e = 0; e < kE; e++) acc[e] += xv * wr[e];
  }
  float v0 = -__builtin_inff(), v1 = -__builtin_inff();
  int e0 = 0, e1 = 0;
#pragma unroll
  for (int e = 0; e < kE; e++) {
    float v = acc[e];
    if (v > v0)      { v1 = v0; e1 = e0; v0 = v; e0 = e; }
    else if (v > v1) { v1 = v; e1 = e; }
  }
  float g0 = 1.f / (1.f + __expf(v1 - v0));
  tok_e[2 * i] = e0; tok_e[2 * i + 1] = e1;
  tok_g[2 * i] = g0; tok_g[2 * i + 1] = 1.f - g0;
  atomicAdd(&counts[e0], 1);
  atomicAdd(&counts[e1], 1);
}

__global__ void moe_offsets(const int* __restrict__ counts, int* __restrict__ offs) {
  if (blockIdx.x == 0 && threadIdx.x == 0) {
    int o = 0;
    for (int e = 0; e < kE; e++) { offs[e] = o; o += (counts[e] + 127) & ~127; }
    offs[kE] = o;
  }
}

__global__ void moe_scatter(const int* __restrict__ tok_e, const float* __restrict__ tok_g,
                            const int* __restrict__ offs, int* __restrict__ fill,
                            int* __restrict__ slot_token, float* __restrict__ slot_gate) {
  int i = blockIdx.x * 256 + threadIdx.x;
  if (i >= kNTOK) return;
#pragma unroll
  for (int j = 0; j < 2; j++) {
    int e = tok_e[2 * i + j];
    int p = atomicAdd(&fill[e], 1);
    int s = offs[e] + p;
    slot_token[s] = i;
    slot_gate[s] = tok_g[2 * i + j];
  }
}

// ---------------- MoE up GEMM: 128x64 block, dual-B (w1,w2) + SwiGLU ----------------

__global__ void gemm_moe_up(const __bf16* __restrict__ X, const __bf16* __restrict__ W1T,
                            const __bf16* __restrict__ W2T, const float* __restrict__ b1,
                            const float* __restrict__ b2, const int* __restrict__ offs,
                            const int* __restrict__ counts, const int* __restrict__ slot_token,
                            __bf16* __restrict__ Hm) {
  int e = blockIdx.z;
  int cnt = counts[e];
  int cntp = (cnt + 127) & ~127;
  int bm0 = blockIdx.y * 128;
  if (bm0 >= cntp) return;
  int off_e = offs[e];
  const __bf16* B1 = W1T + (size_t)e * kHDP * kD;
  const __bf16* B2 = W2T + (size_t)e * kHDP * kD;

  __shared__ alignas(16) __bf16 At[2][128 * 32];
  __shared__ alignas(16) __bf16 B1t[2][64 * 32];
  __shared__ alignas(16) __bf16 B2t[2][64 * 32];
  int tid = threadIdx.x;
  int bn0 = blockIdx.x * 64;
  int lane = tid & 31, w = tid >> 5;
  int wm = (w & 3) * 32, wn = (w >> 2) * 32;
  int ln = lane & 15, hf = lane >> 4;
  int arow = tid >> 1, ak = (tid & 1) * 16;
  int brow = tid >> 2, bks = (tid & 3) * 8;

  int tokA = slot_token[off_e + bm0 + arow];
  const __bf16* Ap = X + (size_t)tokA * kD + ak;   // tokA<0 handled by guard
  const __bf16* B1p = B1 + (size_t)(bn0 + brow) * kD + bks;
  const __bf16* B2p = B2 + (size_t)(bn0 + brow) * kD + bks;

  v8f a1[2][2], a2[2][2];
#pragma unroll
  for (int i = 0; i < 2; i++)
#pragma unroll
    for (int j = 0; j < 2; j++) { a1[i][j] = zero8(); a2[i][j] = zero8(); }

  const int nt = kD / 32;
  v8bf ag0 = (tokA >= 0) ? ldg8(Ap) : zero8bf();
  v8bf ag1 = (tokA >= 0) ? ldg8(Ap + 8) : zero8bf();
  v8bf bg1 = ldg8(B1p), bg2 = ldg8(B2p);
  { __bf16* d0 = At[0] + arow * 32 + ak; *(v8bf*)d0 = ag0; *(v8bf*)(d0 + 8) = ag1;
    *(v8bf*)(B1t[0] + brow * 32 + bks) = bg1;
    *(v8bf*)(B2t[0] + brow * 32 + bks) = bg2; }
  __syncthreads();

  int buf = 0;
  for (int kt = 0; kt < nt; kt++) {
    if (kt + 1 < nt) {
      ag0 = (tokA >= 0) ? ldg8(Ap + (kt + 1) * 32) : zero8bf();
      ag1 = (tokA >= 0) ? ldg8(Ap + (kt + 1) * 32 + 8) : zero8bf();
      bg1 = ldg8(B1p + (kt + 1) * 32);
      bg2 = ldg8(B2p + (kt + 1) * 32);
    }
    const __bf16* Ab = At[buf];
    v16bf af0 = ld_frag(Ab, wm + ln, hf);
    v16bf af1 = ld_frag(Ab, wm + 16 + ln, hf);
#pragma unroll
    for (int j = 0; j < 2; j++) {
      v16bf f1 = ld_frag(B1t[buf], wn + j * 16 + ln, hf);
      a1[0][j] = wmma_bf16(af0, f1, a1[0][j]);
      a1[1][j] = wmma_bf16(af1, f1, a1[1][j]);
      v16bf f2 = ld_frag(B2t[buf], wn + j * 16 + ln, hf);
      a2[0][j] = wmma_bf16(af0, f2, a2[0][j]);
      a2[1][j] = wmma_bf16(af1, f2, a2[1][j]);
    }
    if (kt + 1 < nt) {
      __bf16* d0 = At[buf ^ 1] + arow * 32 + ak; *(v8bf*)d0 = ag0; *(v8bf*)(d0 + 8) = ag1;
      *(v8bf*)(B1t[buf ^ 1] + brow * 32 + bks) = bg1;
      *(v8bf*)(B2t[buf ^ 1] + brow * 32 + bks) = bg2;
    }
    __syncthreads();
    buf ^= 1;
  }

#pragma unroll
  for (int i = 0; i < 2; i++)
#pragma unroll
    for (int j = 0; j < 2; j++)
#pragma unroll
      for (int r = 0; r < 8; r++) {
        int srow = off_e + bm0 + wm + i * 16 + r + 8 * hf;
        int c = bn0 + wn + j * 16 + ln;
        float hv = 0.f;
        if (c < kHD) {
          float u = a1[i][j][r] + b1[(size_t)e * kHD + c];
          float vv = a2[i][j][r] + b2[(size_t)e * kHD + c];
          hv = u * vv / (1.f + __expf(-vv));
        }
        Hm[(size_t)srow * kHDP + c] = (__bf16)hv;
      }
}

// ---------------- MoE down GEMM: 128x128, TDM-fed LDS tiles, gated scatter-add ----------------

__global__ void gemm_moe_down(const __bf16* __restrict__ Hm, const __bf16* __restrict__ W3T,
                              const float* __restrict__ b3, const int* __restrict__ offs,
                              const int* __restrict__ counts, const int* __restrict__ slot_token,
                              const float* __restrict__ slot_gate, float* __restrict__ out) {
  int e = blockIdx.z;
  int cnt = counts[e];
  int cntp = (cnt + 127) & ~127;
  int bm0 = blockIdx.y * 128;
  if (bm0 >= cntp) return;
  int off_e = offs[e];
  const __bf16* Abase = Hm + ((size_t)off_e + bm0) * kHDP;
  const __bf16* Bbase = W3T + (size_t)e * kD * kHDP + (size_t)(blockIdx.x * 128) * kHDP;

  __shared__ alignas(16) __bf16 At[2][128 * 32];
  __shared__ alignas(16) __bf16 Bt[2][128 * 32];
  int tid = threadIdx.x;
  int bn0 = blockIdx.x * 128;
  int lane = tid & 31, w = tid >> 5;
  int wm = (w & 3) * 32, wn = (w >> 2) * 64;
  int ln = lane & 15, hf = lane >> 4;

  unsigned ldsA[2] = { (unsigned)(uintptr_t)&At[0][0], (unsigned)(uintptr_t)&At[1][0] };
  unsigned ldsB[2] = { (unsigned)(uintptr_t)&Bt[0][0], (unsigned)(uintptr_t)&Bt[1][0] };

  v8f acc[2][4];
#pragma unroll
  for (int i = 0; i < 2; i++)
#pragma unroll
    for (int j = 0; j < 4; j++) acc[i][j] = zero8();

  const int nt = kHDP / 32;
  if (tid < 32) {  // wave 0 drives the Tensor Data Mover
    tdm_load_tile(ldsA[0], Abase, 128, kHDP);
    tdm_load_tile(ldsB[0], Bbase, 128, kHDP);
  }

  int buf = 0;
  for (int kt = 0; kt < nt; kt++) {
    __syncthreads();   // all waves done reading the buffer the next DMA overwrites
    if (tid < 32) {
      if (kt + 1 < nt) {
        tdm_load_tile(ldsA[buf ^ 1], Abase + (kt + 1) * 32, 128, kHDP);
        tdm_load_tile(ldsB[buf ^ 1], Bbase + (kt + 1) * 32, 128, kHDP);
        __builtin_amdgcn_s_wait_tensorcnt(2);  // oldest pair (tile kt) complete
      } else {
        __builtin_amdgcn_s_wait_tensorcnt(0);
      }
    }
    __syncthreads();   // tile kt visible to all waves
    const __bf16* Ab = At[buf];
    const __bf16* Bb = Bt[buf];
    v16bf af0 = ld_frag(Ab, wm + ln, hf);
    v16bf af1 = ld_frag(Ab, wm + 16 + ln, hf);
#pragma unroll
    for (int j = 0; j < 4; j++) {
      v16bf bfj = ld_frag(Bb, wn + j * 16 + ln, hf);
      acc[0][j] = wmma_bf16(af0, bfj, acc[0][j]);
      acc[1][j] = wmma_bf16(af1, bfj, acc[1][j]);
    }
    buf ^= 1;
  }

#pragma unroll
  for (int i = 0; i < 2; i++)
#pragma unroll
    for (int r = 0; r < 8; r++) {
      int srow = off_e + bm0 + wm + i * 16 + r + 8 * hf;
      int token = slot_token[srow];
      if (token < 0) continue;
      float g = slot_gate[srow];
#pragma unroll
      for (int j = 0; j < 4; j++) {
        int c = bn0 + wn + j * 16 + ln;
        atomicAdd(&out[(size_t)token * kD + c],
                  g * (acc[i][j][r] + b3[(size_t)e * kD + c]));
      }
    }
}

// ---------------- launch ----------------

extern "C" void kernel_launch(void* const* d_in, const int* in_sizes, int n_in,
                              void* d_out, int out_size, void* d_ws, size_t ws_size,
                              hipStream_t stream) {
  (void)in_sizes; (void)n_in; (void)out_size; (void)ws_size;
  const float* x    = (const float*)d_in[0];
  const float* ln1g = (const float*)d_in[1];
  const float* ln1b = (const float*)d_in[2];
  const float* wq   = (const float*)d_in[3];
  const float* wk   = (const float*)d_in[4];
  const float* wv   = (const float*)d_in[5];
  const float* wo   = (const float*)d_in[6];
  const float* bo   = (const float*)d_in[7];
  const float* ln2g = (const float*)d_in[8];
  const float* ln2b = (const float*)d_in[9];
  const float* rw   = (const float*)d_in[10];
  const float* rb   = (const float*)d_in[11];
  const float* w1   = (const float*)d_in[12];
  const float* b1   = (const float*)d_in[13];
  const float* w2   = (const float*)d_in[14];
  const float* b2   = (const float*)d_in[15];
  const float* w3   = (const float*)d_in[16];
  const float* b3   = (const float*)d_in[17];
  float* out = (float*)d_out;

  char* base = (char*)d_ws;
  size_t off = 0;
  auto take = [&](size_t bytes) -> char* {
    char* r = base + off;
    off += (bytes + 255) & ~(size_t)255;
    return r;
  };
  __bf16* xnb  = (__bf16*)take((size_t)kNTOK * kD * 2);
  __bf16* wqp  = (__bf16*)take((size_t)kD * kD * 2);
  __bf16* wkp  = (__bf16*)take((size_t)kD * kD * 2);
  __bf16* wvp  = (__bf16*)take((size_t)kD * kD * 2);
  __bf16* wop  = (__bf16*)take((size_t)kD * kD * 2);
  __bf16* qbuf = (__bf16*)take((size_t)kNTOK * kD * 2);
  __bf16* kbuf = (__bf16*)take((size_t)kNTOK * kD * 2);
  __bf16* vbufT= (__bf16*)take((size_t)kNTOK * kD * 2);
  __bf16* obuf = (__bf16*)take((size_t)kNTOK * kD * 2);
  __bf16* xn2b = (__bf16*)take((size_t)kNTOK * kD * 2);
  __bf16* w1p  = (__bf16*)take((size_t)kE * kHDP * kD * 2);
  __bf16* w2p  = (__bf16*)take((size_t)kE * kHDP * kD * 2);
  __bf16* w3p  = (__bf16*)take((size_t)kE * kD * kHDP * 2);
  __bf16* hbuf = (__bf16*)take((size_t)kSLOTS * kHDP * 2);
  int*   counts     = (int*)take(kE * 4);
  int*   fill       = (int*)take(kE * 4);
  int*   offs       = (int*)take((kE + 1) * 4);
  int*   tok_e      = (int*)take(2 * kNTOK * 4);
  float* tok_g      = (float*)take(2 * kNTOK * 4);
  int*   slot_token = (int*)take(kSLOTS * 4);
  float* slot_gate  = (float*)take(kSLOTS * 4);

  // weight packing (all transposed to BT[N][K] bf16)
  int nDD = kD * kD;
  pack_headw_t<<<(nDD + 255) / 256, 256, 0, stream>>>(wq, wqp);
  pack_headw_t<<<(nDD + 255) / 256, 256, 0, stream>>>(wk, wkp);
  pack_headw_t<<<(nDD + 255) / 256, 256, 0, stream>>>(wv, wvp);
  pack_t<<<(nDD + 255) / 256, 256, 0, stream>>>(wo, wop);
  size_t nEW = (size_t)kE * kHDP * kD;
  pack_w12_t<<<(unsigned)((nEW + 255) / 256), 256, 0, stream>>>(w1, w1p);
  pack_w12_t<<<(unsigned)((nEW + 255) / 256), 256, 0, stream>>>(w2, w2p);
  pack_w3_t<<<(unsigned)((nEW + 255) / 256), 256, 0, stream>>>(w3, w3p);

  // LN1
  ln_kernel<<<kNTOK, 256, 0, stream>>>(x, ln1g, ln1b, xnb);

  // QKV projections (V stored transposed for attention)
  dim3 gDD(kD / 128, kNTOK / 128);
  gemm128<0><<<gDD, 256, 0, stream>>>(xnb, wqp, qbuf, nullptr, nullptr, kNTOK, kD, kD);
  gemm128<0><<<gDD, 256, 0, stream>>>(xnb, wkp, kbuf, nullptr, nullptr, kNTOK, kD, kD);
  gemm128<2><<<gDD, 256, 0, stream>>>(xnb, wvp, vbufT, nullptr, nullptr, kNTOK, kD, kD);

  // attention
  flash_attn<<<dim3(kT / 128, kBATCH * kH), 256, 0, stream>>>(qbuf, kbuf, vbufT, obuf);

  // output projection + bias + residual -> d_out (= x1)
  gemm128<1><<<gDD, 256, 0, stream>>>(obuf, wop, out, bo, x, kNTOK, kD, kD);

  // LN2 on x1
  ln_kernel<<<kNTOK, 256, 0, stream>>>(out, ln2g, ln2b, xn2b);

  // routing
  moe_zero<<<1, 64, 0, stream>>>(counts, fill);
  slot_init<<<(kSLOTS + 255) / 256, 256, 0, stream>>>(slot_token);
  router_kernel<<<kNTOK / 256, 256, 0, stream>>>(xn2b, rw, rb, tok_e, tok_g, counts);
  moe_offsets<<<1, 32, 0, stream>>>(counts, offs);
  moe_scatter<<<kNTOK / 256, 256, 0, stream>>>(tok_e, tok_g, offs, fill, slot_token, slot_gate);

  // experts
  gemm_moe_up<<<dim3(kHDP / 64, kNTOK / 128, kE), 256, 0, stream>>>(
      xn2b, w1p, w2p, b1, b2, offs, counts, slot_token, hbuf);
  gemm_moe_down<<<dim3(kD / 128, kNTOK / 128, kE), 256, 0, stream>>>(
      hbuf, w3p, b3, offs, counts, slot_token, slot_gate, out);
}
